// BinaryExceptOutliersLinear_53824530153496
// MI455X (gfx1250) — matrix operand order repaired and checked by reading, not compiled, not run on hardware
//
#include <hip/hip_runtime.h>
#include <hip/hip_bf16.h>

typedef __attribute__((ext_vector_type(16))) _Float16 v16h;
typedef __attribute__((ext_vector_type(8)))  float    v8f;

#define BM 128
#define BN 128
#define BK 32
#define LDSS 40   // padded LDS row stride (halves): 80B rows, 16B-aligned sublanes

#if defined(__AMDGCN__) && __has_builtin(__builtin_amdgcn_global_load_async_to_lds_b128) && __has_builtin(__builtin_amdgcn_s_wait_asynccnt)
#define HAVE_ASYNC 1
typedef int async_v4i __attribute__((vector_size(16)));   // matches builtin's parameter type
#else
#define HAVE_ASYNC 0
#endif

// ---------------- stage 1: mean / std(ddof=1) reduction ----------------
__global__ __launch_bounds__(256) void wsum_partial(const float* __restrict__ w, int n,
                                                    double* __restrict__ partials)
{
    __shared__ double ss[256], sq[256];
    const int tid = threadIdx.x;
    double s = 0.0, q = 0.0;
    for (int i = blockIdx.x * 256 + tid; i < n; i += gridDim.x * 256) {
        double v = (double)w[i];
        s += v; q += v * v;
    }
    ss[tid] = s; sq[tid] = q;
    __syncthreads();
    for (int off = 128; off > 0; off >>= 1) {
        if (tid < off) { ss[tid] += ss[tid + off]; sq[tid] += sq[tid + off]; }
        __syncthreads();
    }
    if (tid == 0) { partials[2 * blockIdx.x] = ss[0]; partials[2 * blockIdx.x + 1] = sq[0]; }
}

__global__ __launch_bounds__(256) void wsum_final(const double* __restrict__ partials, int nblk,
                                                  long long n, float* __restrict__ thr)
{
    __shared__ double ss[256], sq[256];
    const int tid = threadIdx.x;
    double s = 0.0, q = 0.0;
    if (tid < nblk) { s = partials[2 * tid]; q = partials[2 * tid + 1]; }
    ss[tid] = s; sq[tid] = q;
    __syncthreads();
    for (int off = 128; off > 0; off >>= 1) {
        if (tid < off) { ss[tid] += ss[tid + off]; sq[tid] += sq[tid + off]; }
        __syncthreads();
    }
    if (tid == 0) {
        double mean = ss[0] / (double)n;
        double var  = (sq[0] - (double)n * mean * mean) / ((double)n - 1.0);
        double sd   = sqrt(var);
        thr[0] = (float)(mean - sd);
        thr[1] = (float)(mean + sd);
    }
}

// ---------------- stage 2: binarize-except-outliers -> f16 ----------------
__global__ __launch_bounds__(256) void binarize_w_f16(const float* __restrict__ w,
                                                      const float* __restrict__ thr,
                                                      _Float16* __restrict__ wh, int n4)
{
    int i = blockIdx.x * 256 + threadIdx.x;
    if (i >= n4) return;
    const float lo = thr[0], hi = thr[1];
    float4 v = ((const float4*)w)[i];
    float e[4] = { v.x, v.y, v.z, v.w };
    union { _Float16 h[4]; uint2 u; } p;
#pragma unroll
    for (int j = 0; j < 4; ++j) {
        float f = e[j];
        float r = (f < lo || f > hi) ? f : (f > 0.f ? 1.f : (f < 0.f ? -1.f : 0.f));
        p.h[j] = (_Float16)r;
    }
    ((uint2*)wh)[i] = p.u;
}

// ---------------- stage 3: x fp32 -> f16 ----------------
__global__ __launch_bounds__(256) void convert_x_f16(const float* __restrict__ x,
                                                     _Float16* __restrict__ xh, int n4)
{
    int i = blockIdx.x * 256 + threadIdx.x;
    if (i >= n4) return;
    float4 v = ((const float4*)x)[i];
    union { _Float16 h[4]; uint2 u; } p;
    p.h[0] = (_Float16)v.x; p.h[1] = (_Float16)v.y;
    p.h[2] = (_Float16)v.z; p.h[3] = (_Float16)v.w;
    ((uint2*)xh)[i] = p.u;
}

// ---------------- stage 4: WMMA GEMM: C = A(MxK) * B(NxK)^T + bias ----------------
__global__ __launch_bounds__(256) void gemm_wmma_f16(const _Float16* __restrict__ A,  // M x K
                                                     const _Float16* __restrict__ B,  // N x K
                                                     const float* __restrict__ bias,
                                                     float* __restrict__ C,           // M x N
                                                     int M, int N, int K)
{
#if HAVE_ASYNC
    __shared__ _Float16 As[3][BM * LDSS];
    __shared__ _Float16 Bs[3][BN * LDSS];
#else
    __shared__ _Float16 As[1][BM * LDSS];
    __shared__ _Float16 Bs[1][BN * LDSS];
#endif

    const int tid  = threadIdx.x;
    const int wave = tid >> 5;
    const int lane = tid & 31;
    const int half = lane >> 4;     // 0/1
    const int l16  = lane & 15;

    const int m0 = blockIdx.y * BM;
    const int n0 = blockIdx.x * BN;

    const int wm = (wave >> 1) << 5;   // 0,32,64,96
    const int wn = (wave & 1) << 6;    // 0,64

    v8f zero = {0.f, 0.f, 0.f, 0.f, 0.f, 0.f, 0.f, 0.f};
    v8f acc[2][4];
#pragma unroll
    for (int i = 0; i < 2; ++i)
#pragma unroll
        for (int j = 0; j < 4; ++j) acc[i][j] = zero;

    // each thread moves two 16B quads for A and two for B per K-step
    const int row0 = tid >> 2;        // 0..63
    const int c8   = (tid & 3) * 8;   // column start within BK
    const int row1 = row0 + 64;

    // per-thread global source pointers (advance by BK per K-step)
    const _Float16* Ag0 = A + (size_t)(m0 + row0) * K + c8;
    const _Float16* Ag1 = A + (size_t)(m0 + row1) * K + c8;
    const _Float16* Bg0 = B + (size_t)(n0 + row0) * K + c8;
    const _Float16* Bg1 = B + (size_t)(n0 + row1) * K + c8;

    const int lofsA0 = row0 * LDSS + c8;
    const int lofsA1 = row1 * LDSS + c8;

    const int KT = K / BK;

#if HAVE_ASYNC
    // ---- CDNA5 async global->LDS, 3-buffer / 2-tile-deep pipeline (ASYNCcnt) ----
#define ISSUE_TILE(kb, buf) do {                                                                                 \
        __builtin_amdgcn_global_load_async_to_lds_b128((async_v4i*)(Ag0 + (kb)), (async_v4i*)(As[buf] + lofsA0), 0, 0); \
        __builtin_amdgcn_global_load_async_to_lds_b128((async_v4i*)(Ag1 + (kb)), (async_v4i*)(As[buf] + lofsA1), 0, 0); \
        __builtin_amdgcn_global_load_async_to_lds_b128((async_v4i*)(Bg0 + (kb)), (async_v4i*)(Bs[buf] + lofsA0), 0, 0); \
        __builtin_amdgcn_global_load_async_to_lds_b128((async_v4i*)(Bg1 + (kb)), (async_v4i*)(Bs[buf] + lofsA1), 0, 0); \
    } while (0)

    // prologue: tiles 0 and 1 in flight
    ISSUE_TILE(0, 0);
    if (KT > 1) {
        ISSUE_TILE(BK, 1);
        __builtin_amdgcn_s_wait_asynccnt(4);   // tile 0 done; tile 1 may stay in flight
    } else {
        __builtin_amdgcn_s_wait_asynccnt(0);
    }
    __syncthreads();

    int bc = 0;  // buffer holding tile kt
    for (int kt = 0; kt < KT; ++kt) {
        const int b2 = (bc >= 1) ? bc - 1 : bc + 2;       // (bc+2) % 3
        if (kt + 2 < KT) {
            // overwrites buffer last read in iteration kt-1 (safe: barrier ended kt-1)
            ISSUE_TILE((size_t)(kt + 2) * BK, b2);
        }

        const _Float16* Ac = As[bc];
        const _Float16* Bc = Bs[bc];

        // A frag (16x32 f16): half 0 -> K{0..7,16..23}, half 1 -> K{8..15,24..31}
        v16h afr[2], bfr[4];
#pragma unroll
        for (int i = 0; i < 2; ++i) {
            const int mrow = wm + i * 16 + l16;
            const uint4* p = (const uint4*)(Ac + mrow * LDSS + half * 8);
            union { uint4 q[2]; v16h v; } u;
            u.q[0] = p[0];
            u.q[1] = p[2];
            afr[i] = u.v;
        }
        // B frag (32x16): half 0 -> K{0..7,16..23}, half 1 -> K{8..15,24..31} of column l16
#pragma unroll
        for (int j = 0; j < 4; ++j) {
            const int nrow = wn + j * 16 + l16;
            const uint4* p = (const uint4*)(Bc + nrow * LDSS + half * 8);
            union { uint4 q[2]; v16h v; } u;
            u.q[0] = p[0];
            u.q[1] = p[2];
            bfr[j] = u.v;
        }

#pragma unroll
        for (int i = 0; i < 2; ++i)
#pragma unroll
            for (int j = 0; j < 4; ++j)
                acc[i][j] = __builtin_amdgcn_wmma_f32_16x16x32_f16(
                    false, afr[i], false, bfr[j], (short)0, acc[i][j], false, false);

        // tile kt+1 must be resident before next iteration; tile kt+2 may stay in flight
        if (kt + 2 < KT) __builtin_amdgcn_s_wait_asynccnt(4);
        else             __builtin_amdgcn_s_wait_asynccnt(0);
        __syncthreads();

        bc = (bc == 2) ? 0 : bc + 1;                      // (bc+1) % 3
    }
#undef ISSUE_TILE
#else
    // ---- fallback: register-staged single-buffer pipeline ----
    uint4 a_st[2], b_st[2];
    a_st[0] = *(const uint4*)Ag0;
    a_st[1] = *(const uint4*)Ag1;
    b_st[0] = *(const uint4*)Bg0;
    b_st[1] = *(const uint4*)Bg1;
    *(uint4*)(As[0] + lofsA0) = a_st[0];
    *(uint4*)(As[0] + lofsA1) = a_st[1];
    *(uint4*)(Bs[0] + lofsA0) = b_st[0];
    *(uint4*)(Bs[0] + lofsA1) = b_st[1];
    __syncthreads();

    for (int kt = 0; kt < KT; ++kt) {
        if (kt + 1 < KT) {
            const size_t kb = (size_t)(kt + 1) * BK;
            a_st[0] = *(const uint4*)(Ag0 + kb);
            a_st[1] = *(const uint4*)(Ag1 + kb);
            b_st[0] = *(const uint4*)(Bg0 + kb);
            b_st[1] = *(const uint4*)(Bg1 + kb);
        }

        v16h afr[2], bfr[4];
#pragma unroll
        for (int i = 0; i < 2; ++i) {
            const int mrow = wm + i * 16 + l16;
            const uint4* p = (const uint4*)(As[0] + mrow * LDSS + half * 8);
            union { uint4 q[2]; v16h v; } u;
            u.q[0] = p[0];
            u.q[1] = p[2];
            afr[i] = u.v;
        }
#pragma unroll
        for (int j = 0; j < 4; ++j) {
            const int nrow = wn + j * 16 + l16;
            const uint4* p = (const uint4*)(Bs[0] + nrow * LDSS + half * 8);
            union { uint4 q[2]; v16h v; } u;
            u.q[0] = p[0];
            u.q[1] = p[2];
            bfr[j] = u.v;
        }

#pragma unroll
        for (int i = 0; i < 2; ++i)
#pragma unroll
            for (int j = 0; j < 4; ++j)
                acc[i][j] = __builtin_amdgcn_wmma_f32_16x16x32_f16(
                    false, afr[i], false, bfr[j], (short)0, acc[i][j], false, false);

        __syncthreads();
        if (kt + 1 < KT) {
            *(uint4*)(As[0] + lofsA0) = a_st[0];
            *(uint4*)(As[0] + lofsA1) = a_st[1];
            *(uint4*)(Bs[0] + lofsA0) = b_st[0];
            *(uint4*)(Bs[0] + lofsA1) = b_st[1];
            __syncthreads();
        }
    }
#endif

    // epilogue: C/D layout — VGPR r: lanes0-15 M=r, lanes16-31 M=8+r, N=l16
#pragma unroll
    for (int j = 0; j < 4; ++j) {
        const int n  = n0 + wn + j * 16 + l16;
        const float bv = bias[n];
#pragma unroll
        for (int i = 0; i < 2; ++i) {
            const int mb = m0 + wm + i * 16 + half * 8;
#pragma unroll
            for (int r = 0; r < 8; ++r)
                C[(size_t)(mb + r) * N + n] = acc[i][j][r] + bv;
        }
    }
}

extern "C" void kernel_launch(void* const* d_in, const int* in_sizes, int n_in,
                              void* d_out, int out_size, void* d_ws, size_t ws_size,
                              hipStream_t stream) {
    const float* x    = (const float*)d_in[0];
    const float* w    = (const float*)d_in[1];
    const float* bias = (const float*)d_in[2];
    float* out = (float*)d_out;

    const int N  = in_sizes[2];              // d_out = 4096
    const int K  = in_sizes[1] / N;          // d_in  = 4096
    const int M  = in_sizes[0] / K;          // B*S   = 16384
    const int nw = in_sizes[1];              // 16.7M weight elements

    // workspace layout
    char* ws = (char*)d_ws;
    _Float16* x_h = (_Float16*)ws;
    size_t x_bytes = (size_t)M * (size_t)K * sizeof(_Float16);
    _Float16* w_h = (_Float16*)(ws + x_bytes);
    size_t w_bytes = (size_t)nw * sizeof(_Float16);
    double* partials = (double*)(ws + x_bytes + w_bytes);
    float*  thr      = (float*)(partials + 2 * 256);

    // 1) mean/std thresholds
    wsum_partial<<<256, 256, 0, stream>>>(w, nw, partials);
    wsum_final<<<1, 256, 0, stream>>>(partials, 256, (long long)nw, thr);

    // 2) binarize-except-outliers -> f16
    int nw4 = nw / 4;
    binarize_w_f16<<<(nw4 + 255) / 256, 256, 0, stream>>>(w, thr, w_h, nw4);

    // 3) x -> f16
    int nx4 = in_sizes[0] / 4;
    convert_x_f16<<<(nx4 + 255) / 256, 256, 0, stream>>>(x, x_h, nx4);

    // 4) WMMA GEMM + bias
    dim3 grid(N / BN, M / BM);
    gemm_wmma_f16<<<grid, 256, 0, stream>>>(x_h, w_h, bias, out, M, N, K);
}
